// Symbolization_1872605741839
// MI455X (gfx1250) — matrix operand order repaired and checked by reading, not compiled, not run on hardware
//
#include <hip/hip_runtime.h>

#define B_ 4096
#define S_ 128
#define K_ 512
#define D_ 64

typedef __bf16 bf16_t;
typedef __attribute__((ext_vector_type(16))) bf16_t v16bf;
typedef __attribute__((ext_vector_type(8)))  float  v8f;
typedef unsigned int u32x4 __attribute__((ext_vector_type(4)));
typedef unsigned int u32x8 __attribute__((ext_vector_type(8)));

// ---------------------------------------------------------------------------
// WMMA helpers (CDNA5 wave32, V_WMMA_F32_16X16X32_BF16)
// ---------------------------------------------------------------------------
static __device__ inline v8f wmma_bf16(v16bf a, v16bf b, v8f c) {
  return __builtin_amdgcn_wmma_f32_16x16x32_bf16(false, a, false, b, (short)0, c,
                                                 false, false);
}

// inline tanh (no ocml call): tanh(x) = sign(x) * (1 - 2t/(1+t)), t=e^-2|x|
static __device__ inline float fast_tanh(float x) {
  float ax = fabsf(x);
  float t  = __expf(-2.0f * ax);
  float y  = 1.0f - 2.0f * t / (1.0f + t);
  return copysignf(y, x);
}

// A fragment: 16x32 bf16 (ISA 7.12.2 16-bit A layout).
// lane l: m = l&15 ; slot i holds k = kbase + (i&7) + (i>=8?16:0) + ((l&16)?8:0)
template <bool RELU, bool FULLK>
static __device__ inline v16bf make_A(const float* tile, int row_stride,
                                      int kbase, int lane) {
  int m  = lane & 15;
  int k8 = (lane & 16) ? 8 : 0;
  v16bf a;
#pragma unroll
  for (int i = 0; i < 16; ++i) {
    float v = 0.0f;
    if (FULLK || i < 8) {  // compile-time: no divergent guards
      int k = kbase + k8 + (i & 7) + ((i & 8) ? 16 : 0);
      v = tile[m * row_stride + k];
      if (RELU) v = fmaxf(v, 0.0f);
    }
    a[i] = (bf16_t)v;
  }
  return a;
}

// ---------------------------------------------------------------------------
// Per-block weight staging into LDS in B-fragment order (bf16, zero-padded K).
// ---------------------------------------------------------------------------
template <int IN, int OUT>
static __device__ inline void stage_w(const float* __restrict__ W, v16bf* dst,
                                      int tid, int nthreads) {
  constexpr int KT = (IN + 31) / 32;
  constexpr int NF = KT * (OUT / 16);
  for (int j = tid; j < NF * 32; j += nthreads) {
    int f = j >> 5, ls = j & 31;
    int nt = f / KT, kt = f - nt * KT;
    int n  = nt * 16 + (ls & 15);
    int kb = kt * 32 + ((ls & 16) ? 8 : 0);
    v16bf b;
#pragma unroll
    for (int i = 0; i < 16; ++i) {
      int   k  = kb + (i & 7) + ((i & 8) ? 16 : 0);
      int   kc = (k < IN) ? k : (IN - 1);
      float v  = W[n * IN + kc];
      b[i] = (bf16_t)((k < IN) ? v : 0.0f);
    }
    dst[j] = b;
  }
}

// One 16-token linear tile using pre-staged weight fragments.
template <int IN_DIM, int OUT_DIM, bool RELU_IN, bool ACCUM>
static __device__ inline void lin_tile(const float* in, int in_rs, float* out,
                                       int out_rs, const v16bf* __restrict__ wf,
                                       const float* __restrict__ bias, int lane) {
  constexpr int KT = (IN_DIM + 31) / 32;
#pragma unroll
  for (int nt = 0; nt < OUT_DIM / 16; ++nt) {
    v8f acc = {};
#pragma unroll
    for (int kt = 0; kt < KT; ++kt) {
      v16bf a;
      if constexpr (IN_DIM == 16)
        a = make_A<RELU_IN, false>(in, in_rs, 0, lane);
      else
        a = make_A<RELU_IN, true>(in, in_rs, kt * 32, lane);
      v16bf b = wf[(nt * KT + kt) * 32 + lane];  // 32B LDS vector load
      acc = wmma_bf16(a, b, acc);
    }
    int   n  = nt * 16 + (lane & 15);
    float bv = bias[n];
#pragma unroll
    for (int r = 0; r < 8; ++r) {
      int   m = r + ((lane & 16) ? 8 : 0);
      float v = acc[r] + bv;
      if (ACCUM) v += out[m * out_rs + n];
      out[m * out_rs + n] = v;
    }
  }
}

template <int DIMV>
static __device__ inline void ln_tanh_row(const float* in, int in_rs, float* out,
                                          int out_rs, const float* __restrict__ g,
                                          const float* __restrict__ bb, int lane) {
  if (lane < 16) {
    float v[DIMV];
    float mean = 0.0f;
#pragma unroll
    for (int j = 0; j < DIMV; ++j) {
      v[j] = in[lane * in_rs + j];
      mean += v[j];
    }
    mean *= (1.0f / (float)DIMV);
    float var = 0.0f;
#pragma unroll
    for (int j = 0; j < DIMV; ++j) {
      float d = v[j] - mean;
      var += d * d;
    }
    var *= (1.0f / (float)DIMV);
    float rs = rsqrtf(var + 1e-5f);
#pragma unroll
    for (int j = 0; j < DIMV; ++j)
      out[lane * out_rs + j] = fast_tanh((v[j] - mean) * rs * g[j] + bb[j]);
  }
}

struct MlpP {
  const float *w1, *b1, *w2, *b2, *w3, *b3;
  const float *g1, *bb1, *g2, *bb2;
  const float *r0w1, *r0b1, *r0w2, *r0b2;
  const float *r1w1, *r1b1, *r1w2, *r1b2;
};

// ---------------------------------------------------------------------------
// Tensor Data Mover: 2D tile (128 rows x 64 f32) global -> LDS (async).
// D# per CDNA5 ISA.
// ---------------------------------------------------------------------------
static __device__ __attribute__((always_inline)) void tdm_load_chunk(
    const float* gsrc, unsigned lds_addr) {
  unsigned long long ga = (unsigned long long)(size_t)gsrc;
  u32x4 g0;
  g0[0] = 1u;                                   // count=1, user descriptor
  g0[1] = lds_addr;                             // lds_addr (bytes)
  g0[2] = (unsigned)ga;                         // global_addr[31:0]
  g0[3] = ((unsigned)(ga >> 32) & 0x01FFFFFFu)  // global_addr[56:32]
          | 0x80000000u;                        // type=2 ("image")
  u32x8 g1;
  g1[0] = 0x20000u;             // data_size=2 (4B); no multicast/pad/iterate
  g1[1] = (unsigned)D_ << 16;   // tensor_dim0 = 64
  g1[2] = 128u << 16;           // tensor_dim1 = 128
  g1[3] = (unsigned)D_ << 16;   // tile_dim0 = 64
  g1[4] = 128u;                 // tile_dim1 = 128, tile_dim2 = 0
  g1[5] = (unsigned)D_;         // tensor_dim0_stride = 64
  g1[6] = 0u;
  g1[7] = 0u;
  asm volatile("tensor_load_to_lds %0, %1" ::"s"(g0), "s"(g1) : "memory");
}

// ---------------------------------------------------------------------------
// Encoder: block = 4 waves x 4 tile-iterations = 256 tokens.
// ---------------------------------------------------------------------------
__global__ __launch_bounds__(128) void encode_kernel(const float* __restrict__ x,
                                                     float* __restrict__ e, MlpP p) {
  __shared__ v16bf wf[704];        // 22 KB weight fragments
  __shared__ float A[4][16 * 32];  // 8 KB
  __shared__ float Bt[4][16 * 64]; // 16 KB
  int tid = threadIdx.x, wave = tid >> 5, lane = tid & 31;
  stage_w<16, 32>(p.w2, wf + 0, tid, 128);
  stage_w<32, 64>(p.w3, wf + 64, tid, 128);
  stage_w<64, 32>(p.r0w1, wf + 192, tid, 128);
  stage_w<32, 64>(p.r0w2, wf + 320, tid, 128);
  stage_w<64, 32>(p.r1w1, wf + 448, tid, 128);
  stage_w<32, 64>(p.r1w2, wf + 576, tid, 128);
  __syncthreads();
  float* a  = A[wave];
  float* bt = Bt[wave];
  for (int it = 0; it < 4; ++it) {
    size_t t0 = (size_t)blockIdx.x * 256 + (size_t)(wave * 4 + it) * 16;
    if (lane < 16) {
      float xv = x[t0 + lane];
      float h[16];
      float mean = 0.0f;
#pragma unroll
      for (int j = 0; j < 16; ++j) {
        h[j] = p.w1[j] * xv + p.b1[j];
        mean += h[j];
      }
      mean *= (1.0f / 16.0f);
      float var = 0.0f;
#pragma unroll
      for (int j = 0; j < 16; ++j) {
        float d = h[j] - mean;
        var += d * d;
      }
      var *= (1.0f / 16.0f);
      float rs = rsqrtf(var + 1e-5f);
#pragma unroll
      for (int j = 0; j < 16; ++j)
        a[lane * 32 + j] = fast_tanh((h[j] - mean) * rs * p.g1[j] + p.bb1[j]);
    }
    __syncthreads();
    lin_tile<16, 32, false, false>(a, 32, bt, 64, wf + 0, p.b2, lane);
    __syncthreads();
    ln_tanh_row<32>(bt, 64, a, 32, p.g2, p.bb2, lane);
    __syncthreads();
    lin_tile<32, 64, false, false>(a, 32, bt, 64, wf + 64, p.b3, lane);
    __syncthreads();
    lin_tile<64, 32, true, false>(bt, 64, a, 32, wf + 192, p.r0b1, lane);
    __syncthreads();
    lin_tile<32, 64, true, true>(a, 32, bt, 64, wf + 320, p.r0b2, lane);
    __syncthreads();
    lin_tile<64, 32, true, false>(bt, 64, a, 32, wf + 448, p.r1b1, lane);
    __syncthreads();
    lin_tile<32, 64, true, true>(a, 32, bt, 64, wf + 576, p.r1b2, lane);
    __syncthreads();
    for (int i = lane; i < 16 * 64; i += 32) {
      int m = i >> 6, d = i & 63;
      e[(t0 + m) * D_ + d] = fmaxf(bt[m * 64 + d], 0.0f);
    }
  }
}

// ---------------------------------------------------------------------------
// VQ with double-buffered TDM chunks + cooperative B-fragment staging.
// block: one s, 128 b-tokens; 4 chunks of 128 codes.
// ---------------------------------------------------------------------------
__global__ __launch_bounds__(256) void vq_kernel(const float* __restrict__ e,
                                                 const float* __restrict__ codebooks,
                                                 int* __restrict__ idx_out,
                                                 float* __restrict__ one_hot,
                                                 float* __restrict__ commitbuf) {
  __shared__ __align__(128) float cbf[2][128 * D_];  // 64 KB (double buffer)
  __shared__ v16bf bfrag[16 * 32];                   // 16 KB B fragments
  __shared__ float c2[128];
  __shared__ float e2t[8][16];

  int s = blockIdx.x >> 5;
  int btile = blockIdx.x & 31;
  int tid = threadIdx.x, wave = tid >> 5, lane = tid & 31;
  int b0 = btile * 128 + wave * 16;

  const float* cbs   = codebooks + (size_t)s * (K_ * D_);
  const float* etile = e + ((size_t)b0 * S_ + s) * D_;
  const int    e_rs  = S_ * D_;

  if (lane < 16) {
    const float* ep = etile + (size_t)lane * e_rs;
    float sum = 0.0f;
#pragma unroll
    for (int d = 0; d < D_; ++d) sum += ep[d] * ep[d];
    e2t[wave][lane] = sum;
  }
  v16bf a0 = make_A<false, true>(etile, e_rs, 0, lane);
  v16bf a1 = make_A<false, true>(etile, e_rs, 32, lane);

  float best[8];
  int   bidx[8];
#pragma unroll
  for (int r = 0; r < 8; ++r) {
    best[r] = 3.4e38f;
    bidx[r] = 0;
  }

  unsigned lds0 = (unsigned)(size_t)&cbf[0][0];
  unsigned lds1 = (unsigned)(size_t)&cbf[1][0];

  if (wave == 0) tdm_load_chunk(cbs, lds0);  // kick off chunk 0

  for (int chunk = 0; chunk < 4; ++chunk) {
    if (wave == 0) {
      if (chunk < 3) {
        // issue next chunk into the other buffer, then wait for the oldest
        tdm_load_chunk(cbs + (size_t)(chunk + 1) * 128 * D_,
                       ((chunk + 1) & 1) ? lds1 : lds0);
        __builtin_amdgcn_s_wait_tensorcnt(1);
      } else {
        __builtin_amdgcn_s_wait_tensorcnt(0);
      }
    }
    __syncthreads();
    const float* cur = cbf[chunk & 1];
    // cooperative: B fragments (bf16) + |c|^2, once per chunk for all 8 waves
    for (int j = tid; j < 16 * 32; j += 256) {
      int f = j >> 5, ls = j & 31;  // f = ct*2 + kt
      int ct = f >> 1, kt = f & 1;
      int n  = ct * 16 + (ls & 15);
      int kb = kt * 32 + ((ls & 16) ? 8 : 0);
      v16bf b;
#pragma unroll
      for (int i = 0; i < 16; ++i) b[i] = (bf16_t)cur[n * D_ + kb + (i & 7) + ((i & 8) ? 16 : 0)];
      bfrag[j] = b;
    }
    if (tid < 128) {
      float sum = 0.0f;
#pragma unroll
      for (int d = 0; d < D_; ++d) {
        float v = cur[tid * D_ + d];
        sum += v * v;
      }
      c2[tid] = sum;
    }
    __syncthreads();
    for (int ct = 0; ct < 8; ++ct) {
      v16bf bb0 = bfrag[(ct * 2 + 0) * 32 + lane];
      v16bf bb1 = bfrag[(ct * 2 + 1) * 32 + lane];
      v8f acc = {};
      acc = wmma_bf16(a0, bb0, acc);
      acc = wmma_bf16(a1, bb1, acc);
      int   n   = ct * 16 + (lane & 15);
      float c2n = c2[n];
      int   gi  = chunk * 128 + n;
#pragma unroll
      for (int r = 0; r < 8; ++r) {
        int   m    = r + ((lane & 16) ? 8 : 0);
        float dist = e2t[wave][m] + c2n - 2.0f * acc[r];
        if (dist < best[r]) {
          best[r] = dist;
          bidx[r] = gi;
        }
      }
    }
    __syncthreads();  // all reads of this chunk done before its buffer is reused
  }

#pragma unroll
  for (int off = 8; off >= 1; off >>= 1) {
#pragma unroll
    for (int r = 0; r < 8; ++r) {
      float od = __shfl_xor(best[r], off, 32);
      int   oi = __shfl_xor(bidx[r], off, 32);
      if (od < best[r] || (od == best[r] && oi < bidx[r])) {
        best[r] = od;
        bidx[r] = oi;
      }
    }
  }
  if ((lane & 15) == 0) {
    int mb = (lane >> 4) * 8;
#pragma unroll
    for (int r = 0; r < 8; ++r) {
      int    m = r + mb;
      size_t t = (size_t)(b0 + m) * S_ + s;
      idx_out[t] = bidx[r];
      one_hot[t * (size_t)K_ + bidx[r]] = 1.0f;
      commitbuf[t] = best[r] * (1.0f / ((float)B_ * S_ * D_));
    }
  }
}

// ---------------------------------------------------------------------------
// Decoder: block = 4 waves x 4 tile-iterations = 256 tokens.
// ---------------------------------------------------------------------------
__global__ __launch_bounds__(128) void decode_kernel(const float* __restrict__ x,
                                                     const float* __restrict__ codebooks,
                                                     const int* __restrict__ idx,
                                                     float* __restrict__ out,
                                                     float* __restrict__ reconsbuf,
                                                     MlpP p) {
  __shared__ v16bf wf[672];
  __shared__ float A[4][16 * 32];
  __shared__ float Bt[4][16 * 64];
  int tid = threadIdx.x, wave = tid >> 5, lane = tid & 31;
  stage_w<64, 32>(p.r0w1, wf + 0, tid, 128);
  stage_w<32, 64>(p.r0w2, wf + 128, tid, 128);
  stage_w<64, 32>(p.r1w1, wf + 256, tid, 128);
  stage_w<32, 64>(p.r1w2, wf + 384, tid, 128);
  stage_w<64, 32>(p.w1, wf + 512, tid, 128);
  stage_w<32, 16>(p.w2, wf + 640, tid, 128);
  __syncthreads();
  float* a  = A[wave];
  float* bt = Bt[wave];
  for (int it = 0; it < 4; ++it) {
    size_t t0 = (size_t)blockIdx.x * 256 + (size_t)(wave * 4 + it) * 16;
    for (int i = lane; i < 16 * 64; i += 32) {
      int    m = i >> 6, d = i & 63;
      size_t t = t0 + m;
      int    ss = (int)(t & (S_ - 1));
      int    k  = idx[t];
      bt[m * 64 + d] = codebooks[((size_t)ss * K_ + k) * D_ + d];
    }
    __syncthreads();
    lin_tile<64, 32, true, false>(bt, 64, a, 32, wf + 0, p.r0b1, lane);
    __syncthreads();
    lin_tile<32, 64, true, true>(a, 32, bt, 64, wf + 128, p.r0b2, lane);
    __syncthreads();
    lin_tile<64, 32, true, false>(bt, 64, a, 32, wf + 256, p.r1b1, lane);
    __syncthreads();
    lin_tile<32, 64, true, true>(a, 32, bt, 64, wf + 384, p.r1b2, lane);
    __syncthreads();
    lin_tile<64, 32, true, false>(bt, 64, a, 32, wf + 512, p.b1, lane);
    __syncthreads();
    ln_tanh_row<32>(a, 32, a, 32, p.g1, p.bb1, lane);
    __syncthreads();
    lin_tile<32, 16, false, false>(a, 32, bt, 64, wf + 640, p.b2, lane);
    __syncthreads();
    ln_tanh_row<16>(bt, 64, bt, 64, p.g2, p.bb2, lane);
    __syncthreads();
    if (lane < 16) {
      float acc = p.b3[0];
#pragma unroll
      for (int j = 0; j < 16; ++j) acc += bt[lane * 64 + j] * p.w3[j];
      size_t t = t0 + lane;
      out[t] = acc;
      float d = acc - x[t];
      reconsbuf[t] = d * d * (1.0f / ((float)B_ * S_));
    }
    __syncthreads();
  }
}

// ---------------------------------------------------------------------------
// one_hot/loss zero fill; deterministic fixed-order loss reductions
// ---------------------------------------------------------------------------
__global__ void zero_tail_kernel(float* __restrict__ base, size_t nvec4,
                                 size_t total) {
  size_t i      = (size_t)blockIdx.x * blockDim.x + threadIdx.x;
  size_t stride = (size_t)gridDim.x * blockDim.x;
  float4 z      = make_float4(0.f, 0.f, 0.f, 0.f);
  for (; i < nvec4; i += stride) ((float4*)base)[i] = z;
  if (blockIdx.x == 0 && threadIdx.x == 0)
    for (size_t j = nvec4 * 4; j < total; ++j) base[j] = 0.0f;
}

__global__ __launch_bounds__(256) void reduce1_kernel(const float* __restrict__ src,
                                                      float* __restrict__ partials) {
  __shared__ float sm[256];
  int tid = threadIdx.x;
  sm[tid] = src[(size_t)blockIdx.x * 256 + tid];
  __syncthreads();
  for (int off = 128; off > 0; off >>= 1) {
    if (tid < off) sm[tid] += sm[tid + off];
    __syncthreads();
  }
  if (tid == 0) partials[blockIdx.x] = sm[0];
}

__global__ __launch_bounds__(256) void reduce2_kernel(const float* __restrict__ partials,
                                                      float* __restrict__ loss) {
  __shared__ float sm[256];
  int   tid = threadIdx.x;
  float s   = 0.0f;
  for (int j = 0; j < 16; ++j) s += partials[tid * 16 + j];
  sm[tid] = s;
  __syncthreads();
  for (int off = 128; off > 0; off >>= 1) {
    if (tid < off) sm[tid] += sm[tid + off];
    __syncthreads();
  }
  if (tid == 0) *loss = sm[0];
}

// ---------------------------------------------------------------------------
// launch
// ---------------------------------------------------------------------------
extern "C" void kernel_launch(void* const* d_in, const int* in_sizes, int n_in,
                              void* d_out, int out_size, void* d_ws, size_t ws_size,
                              hipStream_t stream) {
  (void)in_sizes; (void)n_in; (void)out_size; (void)ws_size;
  const float* x = (const float*)d_in[0];
  MlpP ep = {(const float*)d_in[1],  (const float*)d_in[2],  (const float*)d_in[3],
             (const float*)d_in[4],  (const float*)d_in[5],  (const float*)d_in[6],
             (const float*)d_in[7],  (const float*)d_in[8],  (const float*)d_in[9],
             (const float*)d_in[10], (const float*)d_in[11], (const float*)d_in[12],
             (const float*)d_in[13], (const float*)d_in[14], (const float*)d_in[15],
             (const float*)d_in[16], (const float*)d_in[17], (const float*)d_in[18]};
  MlpP dp = {(const float*)d_in[19], (const float*)d_in[20], (const float*)d_in[21],
             (const float*)d_in[22], (const float*)d_in[23], (const float*)d_in[24],
             (const float*)d_in[25], (const float*)d_in[26], (const float*)d_in[27],
             (const float*)d_in[28], (const float*)d_in[29], (const float*)d_in[30],
             (const float*)d_in[31], (const float*)d_in[32], (const float*)d_in[33],
             (const float*)d_in[34], (const float*)d_in[35], (const float*)d_in[36]};
  const float* codebooks = (const float*)d_in[37];

  const size_t NT = (size_t)B_ * S_;
  float* out     = (float*)d_out;
  float* loss    = out + NT;
  float* one_hot = loss + 1;

  char*  ws       = (char*)d_ws;
  float* e        = (float*)ws;
  int*   idx      = (int*)(ws + NT * D_ * sizeof(float));
  float* lossbuf  = (float*)(ws + NT * D_ * sizeof(float) + NT * sizeof(int));
  float* commit   = lossbuf;
  float* recons   = lossbuf + NT;
  float* partials = lossbuf + 2 * NT;

  size_t tail = 1 + NT * (size_t)K_;
  zero_tail_kernel<<<2048, 256, 0, stream>>>(loss, tail / 4, tail);
  encode_kernel<<<(int)(NT / 256), 128, 0, stream>>>(x, e, ep);
  vq_kernel<<<S_ * 32, 256, 0, stream>>>(e, codebooks, idx, one_hot, commit);
  decode_kernel<<<(int)(NT / 256), 128, 0, stream>>>(x, codebooks, idx, out, recons, dp);
  reduce1_kernel<<<(int)(2 * NT / 256), 256, 0, stream>>>(lossbuf, partials);
  reduce2_kernel<<<1, 256, 0, stream>>>(partials, loss);
}
